// PersistentMemory_39685497815916
// MI455X (gfx1250) — compile-verified, matrix operand order checked by reading, not run on hardware
//
#include <hip/hip_runtime.h>
#include <stdint.h>

typedef __attribute__((ext_vector_type(16))) __bf16 v16bf;
typedef __attribute__((ext_vector_type(8)))  float  v8f;

#define NQ      256
#define DIM     512
#define NBANK   500000
#define TOPK    10
#define NCHUNK  250
#define CHUNK   2000          // bank rows per chunk (multiple of 16)
#define TILES   (CHUNK / 16)  // 125
#define KSTEPS  (DIM / 32)    // 16

union BF16x16 {
    v16bf    v;
    uint16_t s[16];
    uint4    q[2];
};

// f32 -> bf16 hi (truncate) + bf16 lo (residual, truncate): ~16 mantissa bits total
__device__ __forceinline__ void f32_to_bf16x2(float a, uint16_t& hi, uint16_t& lo) {
    uint32_t ua = __float_as_uint(a);
    hi = (uint16_t)(ua >> 16);
    float hif = __uint_as_float(ua & 0xffff0000u);
    float lof = a - hif;
    lo = (uint16_t)(__float_as_uint(lof) >> 16);
}

// ---------------- Kernel 1: split queries into bf16 hi/lo planes ----------------
__global__ void pm_convert_q(const float* __restrict__ q,
                             uint16_t* __restrict__ qhi,
                             uint16_t* __restrict__ qlo) {
    int i = blockIdx.x * blockDim.x + threadIdx.x;   // exactly NQ*DIM threads
    uint16_t h, l;
    f32_to_bf16x2(q[i], h, l);
    qhi[i] = h;
    qlo[i] = l;
}

// Convert 8 prefetched float4s and scatter into one B-tile LDS buffer
// buf[0] = hi plane, buf[1] = lo plane, each [KSTEPS][32][16] ushorts
__device__ __forceinline__ void pm_stage_store(const float4 gv[8],
                                               uint16_t (*buf)[KSTEPS][32][16],
                                               int tid) {
    #pragma unroll
    for (int it = 0; it < 8; ++it) {
        int ff = tid + it * 256;            // float4 index 0..2047
        int r  = ff >> 7;                   // bank row within tile (= B column)
        int kk = (ff & 127) * 4;            // K offset, multiple of 4
        float4 v = gv[it];
        uint16_t hs0, hs1, hs2, hs3, ls0, ls1, ls2, ls3;
        f32_to_bf16x2(v.x, hs0, ls0);
        f32_to_bf16x2(v.y, hs1, ls1);
        f32_to_bf16x2(v.z, hs2, ls2);
        f32_to_bf16x2(v.w, hs3, ls3);
        int k32 = kk & 31;
        int ksx = kk >> 5;
        int hh  = (k32 & 15) >> 3;
        int e   = (k32 & 7) | ((k32 & 16) ? 8 : 0);   // elem base (0,4,8,12)
        int ln  = hh * 16 + r;
        unsigned long long H = (unsigned long long)hs0 |
                               ((unsigned long long)hs1 << 16) |
                               ((unsigned long long)hs2 << 32) |
                               ((unsigned long long)hs3 << 48);
        unsigned long long L = (unsigned long long)ls0 |
                               ((unsigned long long)ls1 << 16) |
                               ((unsigned long long)ls2 << 32) |
                               ((unsigned long long)ls3 << 48);
        *(unsigned long long*)(&buf[0][ksx][ln][e]) = H;
        *(unsigned long long*)(&buf[1][ksx][ln][e]) = L;
    }
}

// ---------------- Kernel 2: WMMA GEMM + fused per-chunk top-k ----------------
// Grid: 2*NCHUNK blocks of 256 threads (8 waves). chunk = bid>>1, query-half = bid&1.
// Wave w owns M-tile (qhalf*8 + w): 16 queries, A (hi+lo bf16, full K) resident in VGPRs.
// B tiles double-buffered in LDS; next tile's global loads issued before the WMMA block.
__launch_bounds__(256)
__global__ void pm_gemm_topk(const float* __restrict__ bank,
                             const uint16_t* __restrict__ qhi,
                             const uint16_t* __restrict__ qlo,
                             float*    __restrict__ pscore,
                             uint32_t* __restrict__ pidx) {
    // [buf][hi/lo][kstep][lane][elem] : 64 KB, double-buffered WMMA-B lane image
    __shared__ __align__(16) uint16_t ldsB[2][2][KSTEPS][32][16];
    __shared__ __align__(16) float    ldsScr[8][8][32];          // 8 KB transpose scratch

    const int tid   = threadIdx.x;
    const int lane  = tid & 31;
    const int wave  = tid >> 5;                 // 0..7
    const int chunk = blockIdx.x >> 1;
    const int qhalf = blockIdx.x & 1;
    const int mtile = qhalf * 8 + wave;

    // ---- Load A (16 queries x 512, hi/lo) into registers, ISA 16-bit A layout ----
    const int mrow = mtile * 16 + (lane & 15);
    const int h    = lane >> 4;                 // lane half selects K sub-chunks
    BF16x16 Ahi[KSTEPS], Alo[KSTEPS];
    #pragma unroll
    for (int ks = 0; ks < KSTEPS; ++ks) {
        const uint16_t* ph = qhi + mrow * DIM + ks * 32 + 8 * h;
        const uint16_t* pl = qlo + mrow * DIM + ks * 32 + 8 * h;
        Ahi[ks].q[0] = *(const uint4*)(ph);        // K = 8h + 0..7
        Ahi[ks].q[1] = *(const uint4*)(ph + 16);   // K = 16 + 8h + 0..7
        Alo[ks].q[0] = *(const uint4*)(pl);
        Alo[ks].q[1] = *(const uint4*)(pl + 16);
    }

    // running top-k: lane q (<16) owns query mtile*16+q
    float    ts[TOPK];
    uint32_t ti[TOPK];
    #pragma unroll
    for (int j = 0; j < TOPK; ++j) { ts[j] = -__builtin_inff(); ti[j] = 0u; }

    const int nchunk0 = chunk * CHUNK;
    const int ldr     = tid >> 7;               // this thread's row base (ff>>7)
    const int ldk     = (tid & 127) * 4;        // this thread's K base

    // ---- prologue: stage tile 0 into buffer 0 ----
    {
        float4 gv[8];
        #pragma unroll
        for (int it = 0; it < 8; ++it) {
            int ff = tid + it * 256;
            int r  = ff >> 7;
            int kk = (ff & 127) * 4;
            gv[it] = *(const float4*)(bank + (long)(nchunk0 + r) * DIM + kk);
        }
        pm_stage_store(gv, ldsB[0], tid);
    }
    __syncthreads();

    for (int tile = 0; tile < TILES; ++tile) {
        const int n0  = nchunk0 + tile * 16;
        const int cur = tile & 1;
        const int nxt = cur ^ 1;
        const bool more = (tile + 1) < TILES;

        // ---- issue next tile's global loads first (overlap HBM with WMMA) ----
        float4 gv[8];
        if (more) {
            #pragma unroll
            for (int it = 0; it < 8; ++it) {
                int ff = tid + it * 256;
                int r  = ff >> 7;
                int kk = (ff & 127) * 4;
                gv[it] = *(const float4*)(bank + (long)(n0 + 16 + r) * DIM + kk);
            }
        }

        // ---- 16x16 score tile: 3 independent accumulator chains ----
        v8f acc0 = {0.f, 0.f, 0.f, 0.f, 0.f, 0.f, 0.f, 0.f};
        v8f acc1 = {0.f, 0.f, 0.f, 0.f, 0.f, 0.f, 0.f, 0.f};
        v8f acc2 = {0.f, 0.f, 0.f, 0.f, 0.f, 0.f, 0.f, 0.f};
        #pragma unroll
        for (int ks = 0; ks < KSTEPS; ++ks) {
            BF16x16 Bh, Bl;
            const uint4* sh = (const uint4*)&ldsB[cur][0][ks][lane][0];
            const uint4* sl = (const uint4*)&ldsB[cur][1][ks][lane][0];
            Bh.q[0] = sh[0]; Bh.q[1] = sh[1];
            Bl.q[0] = sl[0]; Bl.q[1] = sl[1];
            acc0 = __builtin_amdgcn_wmma_f32_16x16x32_bf16(false, Ahi[ks].v, false, Bh.v,
                                                           (short)0, acc0, false, false);
            acc1 = __builtin_amdgcn_wmma_f32_16x16x32_bf16(false, Ahi[ks].v, false, Bl.v,
                                                           (short)0, acc1, false, false);
            acc2 = __builtin_amdgcn_wmma_f32_16x16x32_bf16(false, Alo[ks].v, false, Bh.v,
                                                           (short)0, acc2, false, false);
        }

        // ---- convert + store next tile into the other buffer ----
        if (more) pm_stage_store(gv, ldsB[nxt], tid);

        // ---- transpose scores via LDS, update per-query top-k ----
        #pragma unroll
        for (int v = 0; v < 8; ++v)
            ldsScr[wave][v][lane] = acc0[v] + acc1[v] + acc2[v];
        __syncthreads();

        if (lane < 16) {
            // row m=lane: m<8 -> VGPR m lanes 0-15 ; m>=8 -> VGPR m-8 lanes 16-31
            const float* rowS = &ldsScr[wave][lane & 7][(lane >> 3) * 16];
            #pragma unroll
            for (int c = 0; c < 16; ++c) {
                float s = rowS[c];
                if (s > ts[0]) {                      // common case: single compare
                    float    cs = s;
                    uint32_t ci = (uint32_t)(n0 + c);
                    #pragma unroll
                    for (int j = 0; j < TOPK - 1; ++j) {   // branch-free sorted insert
                        float    ns = ts[j + 1];
                        uint32_t ni = ti[j + 1];
                        bool up = cs > ns;
                        ts[j] = up ? ns : cs;
                        ti[j] = up ? ni : ci;
                        cs    = up ? cs : ns;
                        ci    = up ? ci : ni;
                    }
                    ts[TOPK - 1] = cs;
                    ti[TOPK - 1] = ci;
                }
            }
        }
        __syncthreads();   // scr reads done before next overwrite; buf[nxt] visible
    }

    // ---- write per-chunk partial top-k (descending) ----
    if (lane < 16) {
        int  q    = mtile * 16 + lane;
        long base = ((long)chunk * NQ + q) * TOPK;
        #pragma unroll
        for (int j = 0; j < TOPK; ++j) {
            pscore[base + j] = ts[TOPK - 1 - j];
            pidx[base + j]   = ti[TOPK - 1 - j];
        }
    }
    (void)ldr; (void)ldk;
}

// ---------------- Kernel 3: per-query merge of 250x10 partials ----------------
#define NCAND (NCHUNK * TOPK)   // 2500
__global__ void pm_merge_topk(const float* __restrict__ pscore,
                              const uint32_t* __restrict__ pidx,
                              float*    __restrict__ outScores,
                              uint32_t* __restrict__ fidx) {
    __shared__ float    cs_[NCAND];
    __shared__ uint32_t ci_[NCAND];
    __shared__ float    rs[256];
    __shared__ uint32_t ri[256];
    __shared__ int      rslot[256];
    const int q = blockIdx.x, t = threadIdx.x;

    for (int i = t; i < NCAND; i += 256) {
        int  chunkI = i / TOPK, j = i % TOPK;
        long src = ((long)chunkI * NQ + q) * TOPK + j;
        cs_[i] = pscore[src];
        ci_[i] = pidx[src];
    }
    __syncthreads();

    for (int r = 0; r < TOPK; ++r) {
        float    bs = -__builtin_inff();
        uint32_t bi = 0xffffffffu;
        int      bslot = 0;
        for (int i = t; i < NCAND; i += 256) {
            float s = cs_[i]; uint32_t ix = ci_[i];
            if (s > bs || (s == bs && ix < bi)) { bs = s; bi = ix; bslot = i; }
        }
        rs[t] = bs; ri[t] = bi; rslot[t] = bslot;
        __syncthreads();
        for (int stride = 128; stride > 0; stride >>= 1) {
            if (t < stride) {
                float s2 = rs[t + stride]; uint32_t i2 = ri[t + stride];
                if (s2 > rs[t] || (s2 == rs[t] && i2 < ri[t])) {
                    rs[t] = s2; ri[t] = i2; rslot[t] = rslot[t + stride];
                }
            }
            __syncthreads();
        }
        if (t == 0) {
            outScores[q * TOPK + r] = rs[0];
            fidx[q * TOPK + r]      = ri[0];
            cs_[rslot[0]]           = -__builtin_inff();   // claim winner
        }
        __syncthreads();
    }
}

// ---------------- Kernel 4: gather winning bank rows ----------------
__global__ void pm_gather(const float* __restrict__ bank,
                          const uint32_t* __restrict__ fidx,
                          float* __restrict__ outRet) {
    const int g = blockIdx.x;                    // 0..NQ*TOPK-1
    const uint32_t row = fidx[g];
    const float4* src = (const float4*)(bank + (long)row * DIM);
    float4*       dst = (float4*)(outRet + (long)g * DIM);
    dst[threadIdx.x] = src[threadIdx.x];         // 128 threads x float4 = 512 floats
}

extern "C" void kernel_launch(void* const* d_in, const int* in_sizes, int n_in,
                              void* d_out, int out_size, void* d_ws, size_t ws_size,
                              hipStream_t stream) {
    (void)in_sizes; (void)n_in; (void)out_size; (void)ws_size;
    const float* query = (const float*)d_in[0];   // [256,512] f32
    const float* bank  = (const float*)d_in[1];   // [500000,512] f32
    // d_in[2] is k == 10 (compile-time TOPK)

    char* ws = (char*)d_ws;
    uint16_t* qhi    = (uint16_t*)(ws);                               // 256 KB
    uint16_t* qlo    = (uint16_t*)(ws + 262144);                      // 256 KB
    float*    pscore = (float*)   (ws + 524288);                      // 2.56 MB
    uint32_t* pidx   = (uint32_t*)(ws + 524288 + 2560000);            // 2.56 MB
    uint32_t* fidx   = (uint32_t*)(ws + 524288 + 2 * 2560000);        // 10 KB

    float* outRet    = (float*)d_out;                                 // [256][10][512]
    float* outScores = outRet + (long)NQ * TOPK * DIM;                // [256][10]

    pm_convert_q<<<(NQ * DIM) / 256, 256, 0, stream>>>(query, qhi, qlo);
    pm_gemm_topk<<<2 * NCHUNK, 256, 0, stream>>>(bank, qhi, qlo, pscore, pidx);
    pm_merge_topk<<<NQ, 256, 0, stream>>>(pscore, pidx, outScores, fidx);
    pm_gather<<<NQ * TOPK, 128, 0, stream>>>(bank, fidx, outRet);
}
